// RIMNetwork_specialized_68736656605804
// MI455X (gfx1250) — matrix operand order repaired and checked by reading, not compiled, not run on hardware
//
#include <hip/hip_runtime.h>
#include <hip/hip_bf16.h>

// ---------------------------------------------------------------------------
// RIM recurrent network, persistent-kernel design for MI455X (gfx1250).
//
//  * 2 workgroups x 16 batches, 1024 threads (32 wave32) each; batches are
//    independent -> zero inter-workgroup sync across all 512 serial steps.
//  * Entire recurrent state H (16x6x256 f32) lives in LDS for all 512 steps.
//  * All GEMMs on v_wmma_f32_16x16x32_bf16 with hi/lo split-precision
//    compensation (3 WMMAs per K-slab) to retain ~fp32 accuracy.
//  * A-operands staged in LDS as pre-split bf16 hi/lo in WMMA-A element
//    order: each operand = two 16B ds loads, no per-tile cvt VALU.
//  * Weights pre-split to bf16 hi/lo WMMA-B tiles in d_ws by a prep kernel,
//    reused for all 512 steps (L2 resident); stream-out uses NT stores.
// ---------------------------------------------------------------------------

typedef __attribute__((ext_vector_type(16))) __bf16 v16bf;
typedef __attribute__((ext_vector_type(8)))  __bf16 v8bf;
typedef __attribute__((ext_vector_type(8)))  float  v8f;

#define B_    32
#define T_    512
#define I_    96
#define H_    256
#define M_    6
#define R_    16
#define AK_   16
#define OUT_  64
#define BPW_  16      // batches per workgroup
#define NTHREADS_ 1024

// d_out flat offsets (floats): outputs, Hseq, Ain_seq, Acomm_seq, attn_seq
#define OUTS_OFF  0L
#define HSEQ_OFF  1048576L
#define AIN_OFF   26214400L
#define ACOMM_OFF 51380224L
#define ATTN_OFF  76546048L

// ws tile slots: each slot = 1024 bf16 (512 hi then 512 lo), tile = 32K x 16N,
// element (k,n) at lane n+16*((k>>3)&1), elem (k&7)+8*(k>>4)  ->  lane reads its 16 bf16 (32B).
#define SLOT_WIP   0      // 6m x 8kt x 16nt   (m*128 + kt*16 + nt)
#define SLOT_WV    768
#define SLOT_WQ    1536   // 6m x 8kt
#define SLOT_WK    1584
#define SLOT_VR    1632   // 6m x 16nt (K=16 real, rows 16..31 zero)
#define SLOT_WOUT  1728   // 16kt x 4nt
#define N_SLOTS    1792

// LDS layout (bytes)
#define LDS_H     0         // f32  [16b][6m][256h]  state/H_in/H_new   98304
#define LDS_UV    98304     // union: Vc f32 [16b][6m][256h] (98304)
                            //      / A_in staged bf16 hi(32768)+lo(32768)
#define LDS_HH    196608    // bf16 [6m][16b][256h]  H_in hi           49152
#define LDS_HL    245760    // bf16 [6m][16b][256h]  H_in lo           49152
#define LDS_REC   294912    // f32  [16b][6m][16r]                      6144
#define LDS_QC    301056    // f32  [16b][6m][16k]                      6144
#define LDS_KC    307200    // f32  [16b][6m][16k]                      6144
#define LDS_ATT   313344    // f32  [16b][6m][6n]                       2304
#define SMEM_BYTES 315648   // < 320KB WGP limit

// ---------------------------------------------------------------------------
// Prep: split weights into bf16 hi/lo WMMA-B tiles.
// grid = N_SLOTS blocks x 512 threads; thread = (k,n) of one 32x16 tile.
// ---------------------------------------------------------------------------
__global__ void rim_prep_kernel(const float* __restrict__ W_ip,
                                const float* __restrict__ W_Q,
                                const float* __restrict__ W_K,
                                const float* __restrict__ W_V,
                                const float* __restrict__ Vr,
                                const float* __restrict__ W_out,
                                __bf16* __restrict__ ws) {
  const int sid = blockIdx.x;
  const int tid = threadIdx.x;          // 0..511
  const int k = tid >> 4, n = tid & 15;
  float v = 0.f;
  if (sid < SLOT_WV) {                  // W_ip (M,H,H)
    int m = sid >> 7, kt = (sid >> 4) & 7, nt = sid & 15;
    v = W_ip[(m * 256 + kt * 32 + k) * 256 + nt * 16 + n];
  } else if (sid < SLOT_WQ) {           // W_V (M,H,H)
    int s = sid - SLOT_WV;
    int m = s >> 7, kt = (s >> 4) & 7, nt = s & 15;
    v = W_V[(m * 256 + kt * 32 + k) * 256 + nt * 16 + n];
  } else if (sid < SLOT_WK) {           // W_Q (M,H,AK)
    int s = sid - SLOT_WQ;
    int m = s >> 3, kt = s & 7;
    v = W_Q[(m * 256 + kt * 32 + k) * 16 + n];
  } else if (sid < SLOT_VR) {           // W_K (M,H,AK)
    int s = sid - SLOT_WK;
    int m = s >> 3, kt = s & 7;
    v = W_K[(m * 256 + kt * 32 + k) * 16 + n];
  } else if (sid < SLOT_WOUT) {         // Vr (M,R,H), K padded 16 -> 32
    int s = sid - SLOT_VR;
    int m = s >> 4, nt = s & 15;
    v = (k < 16) ? Vr[(m * 16 + k) * 256 + nt * 16 + n] : 0.f;
  } else {                              // W_out (512, 64)
    int s = sid - SLOT_WOUT;
    int kt = s >> 2, nt = s & 3;
    v = W_out[(kt * 32 + k) * 64 + nt * 16 + n];
  }
  __bf16 hi = (__bf16)v;
  long base = (long)sid * 1024;
  int pos = ((n + 16 * ((k >> 3) & 1)) << 4) + (k & 7) + 8 * (k >> 4);
  ws[base + pos]       = hi;
  ws[base + 512 + pos] = (__bf16)(v - (float)hi);
}

// ---------------------------------------------------------------------------
// WMMA helpers
// ---------------------------------------------------------------------------
__device__ __forceinline__ void load_b_tile(const __bf16* __restrict__ ws,
                                            int slot, int lane,
                                            v16bf& bh, v16bf& bl) {
  const __bf16* p = ws + (long)slot * 1024 + lane * 16;
  bh = *(const v16bf*)p;          // lane = N col (+16*half), 16 K values (32B)
  bl = *(const v16bf*)(p + 512);
}

// A operand from bf16 staging laid out row-major [row][K] (K contiguous).
// ISA 16-bit A layout: lane -> M=row=lane%16, half=lane/16; the lane's 16
// elements are two contiguous K-runs of 8:  K = 8*half + 0..7  and
// K = 16 + 8*half + 0..7  ->  two 16B LDS loads + shuffle, zero cvt VALU.
__device__ __forceinline__ v16bf load_a_stage(const __bf16* __restrict__ S,
                                              int base0) {
  v8bf a = *(const v8bf*)(S + base0);
  v8bf b = *(const v8bf*)(S + base0 + 16);
  return __builtin_shufflevector(a, b, 0, 1, 2, 3, 4, 5, 6, 7,
                                       8, 9, 10, 11, 12, 13, 14, 15);
}

// Fallback: build A operand hi/lo from an fp32 fetch(row,k) (small cases).
template <typename F>
__device__ __forceinline__ void gather_a(int lane, F&& fetch,
                                         v16bf& ah, v16bf& al) {
  const int half = lane >> 4;
  const int row  = lane & 15;
#pragma unroll
  for (int j = 0; j < 16; ++j) {
    const int kk = (j < 8) ? (8 * half + j) : (16 + 8 * half + (j - 8));
    float x = fetch(row, kk);
    __bf16 hx = (__bf16)x;
    ah[j] = hx;
    al[j] = (__bf16)(x - (float)hx);
  }
}

// Split-precision MAC: acc += Ah*Bh + Ah*Bl + Al*Bh  (~fp32 accuracy)
__device__ __forceinline__ v8f wmma3(v8f acc, const v16bf& ah, const v16bf& al,
                                     const v16bf& bh, const v16bf& bl) {
  acc = __builtin_amdgcn_wmma_f32_16x16x32_bf16(false, ah, false, bh,
                                                (short)0, acc, false, false);
  acc = __builtin_amdgcn_wmma_f32_16x16x32_bf16(false, ah, false, bl,
                                                (short)0, acc, false, false);
  acc = __builtin_amdgcn_wmma_f32_16x16x32_bf16(false, al, false, bh,
                                                (short)0, acc, false, false);
  return acc;
}

// ---------------------------------------------------------------------------
// Main persistent kernel: grid = 2 workgroups x 1024 threads.
// ---------------------------------------------------------------------------
__global__ __launch_bounds__(NTHREADS_)
void rim_main_kernel(const float* __restrict__ x,
                     const float* __restrict__ U,
                     const float* __restrict__ b_ip,
                     const float* __restrict__ cell_bias,
                     const float* __restrict__ Wtask,
                     const float* __restrict__ Wsens,
                     const float* __restrict__ b_out,
                     const __bf16* __restrict__ ws,
                     float* __restrict__ out) {
  const int tid  = threadIdx.x;
  const int wave = tid >> 5;
  const int lane = tid & 31;
  const int wg   = blockIdx.x;        // 0..1, 16 batches each
  const int half = lane >> 4;
  const int nIdx = lane & 15;
  const int row  = lane & 15;

  extern __shared__ char smem[];
  float*  sH   = (float*)(smem + LDS_H);    // [b][m][h] state / H_in / H_new
  float*  sVc  = (float*)(smem + LDS_UV);   // Vc [b][m][h]
  __bf16* sAh  = (__bf16*)(smem + LDS_UV);  // A_in staged hi [m<4][b][h]
  __bf16* sAl  = sAh + 4 * BPW_ * H_;       // A_in staged lo
  __bf16* sHh  = (__bf16*)(smem + LDS_HH);  // H_in staged hi [m][b][h]
  __bf16* sHl  = (__bf16*)(smem + LDS_HL);  // H_in staged lo
  float*  sRec = (float*)(smem + LDS_REC);  // Hs@U  [b][m][r]
  float*  sQc  = (float*)(smem + LDS_QC);   // [b][m][k]
  float*  sKc  = (float*)(smem + LDS_KC);   // [b][m][k]
  float*  sAtt = (float*)(smem + LDS_ATT);  // [b][m][n]

  for (int i = tid; i < BPW_ * M_ * H_; i += NTHREADS_) sH[i] = 0.f;
  __syncthreads();

#pragma unroll 1
  for (int t = 0; t < T_; ++t) {
    // ---------------- Phase A (VALU): rec-tmp, A_in ----------------
    for (int i = tid; i < BPW_ * M_ * R_; i += NTHREADS_) {   // Hs @ U
      int b = i / (M_ * R_), m = (i / R_) % M_, r = i & 15;
      const float* hrow = sH + (b * M_ + m) * H_;
      const float* ucol = U + (long)m * H_ * R_ + r;
      float s = 0.f;
      for (int h = 0; h < H_; ++h) s += hrow[h] * ucol[h * R_];
      sRec[(b * M_ + m) * R_ + r] = s;
    }
    for (int i = tid; i < 4 * BPW_ * H_; i += NTHREADS_) {    // A_in (m=0..3)
      int m = i >> 12, b = (i >> 8) & 15, h = i & 255;
      int bg = wg * BPW_ + b;
      const float* xr = x + ((long)bg * T_ + t) * I_;
      float s = 0.f;
      if (m < 2) {
        const float* w = Wtask + (m * 32) * 256 + h;
        for (int j = 0; j < 32; ++j) s += xr[64 + j] * w[j * 256];
      } else {
        const float* w = Wsens + ((m - 2) * 64) * 256 + h;
        for (int j = 0; j < 64; ++j) s += xr[j] * w[j * 256];
      }
      __bf16 hi = (__bf16)s;                    // split once, reuse 16x
      sAh[(m * BPW_ + b) * H_ + h] = hi;
      sAl[(m * BPW_ + b) * H_ + h] = (__bf16)(s - (float)hi);
      __builtin_nontemporal_store(
          s, &out[AIN_OFF + (((long)t * B_ + bg) * M_ + m) * H_ + h]);
    }
    for (int i = tid; i < 2 * BPW_ * H_; i += NTHREADS_) {    // A_in zeros m=4,5
      int m = 4 + (i >> 12), b = (i >> 8) & 15, h = i & 255;
      __builtin_nontemporal_store(
          0.f, &out[AIN_OFF + (((long)t * B_ + wg * BPW_ + b) * M_ + m) * H_ + h]);
    }
    __syncthreads();

    // ------- Phase B (WMMA): H_in = tanh(A_in@W_ip + rec@Vr + biases) -------
    for (int tile = wave; tile < M_ * 16; tile += 32) {
      int m = tile >> 4, nt = tile & 15;
      v8f acc = {0.f, 0.f, 0.f, 0.f, 0.f, 0.f, 0.f, 0.f};
      v16bf ah, al, bh, bl;
      if (m < 4) {
        const int abase = (m * BPW_ + row) * H_ + 8 * half;
        for (int kt = 0; kt < 8; ++kt) {
          __builtin_prefetch(
              ws + (long)(SLOT_WIP + m * 128 + ((kt + 1) & 7) * 16 + nt) * 1024,
              0, 0);
          ah = load_a_stage(sAh, abase + kt * 32);
          al = load_a_stage(sAl, abase + kt * 32);
          load_b_tile(ws, SLOT_WIP + m * 128 + kt * 16 + nt, lane, bh, bl);
          acc = wmma3(acc, ah, al, bh, bl);
        }
      }
      // low-rank recurrence: (Hs@U) @ Vr, K=16 padded to 32
      gather_a(lane, [&](int rw, int kk) {
        return (kk < 16) ? sRec[(rw * M_ + m) * R_ + kk] : 0.f; }, ah, al);
      load_b_tile(ws, SLOT_VR + m * 16 + nt, lane, bh, bl);
      acc = wmma3(acc, ah, al, bh, bl);

      int hcol = nt * 16 + nIdx;
      float bias = b_ip[m * H_ + hcol] + cell_bias[m * H_ + hcol];
#pragma unroll
      for (int r = 0; r < 8; ++r) {
        int b = r + 8 * half;
        float v = tanhf(acc[r] + bias);
        sH[(b * M_ + m) * H_ + hcol] = v;
        __bf16 hv = (__bf16)v;                  // split once, reuse 18x
        sHh[(m * BPW_ + b) * H_ + hcol] = hv;
        sHl[(m * BPW_ + b) * H_ + hcol] = (__bf16)(v - (float)hv);
      }
    }
    __syncthreads();

    // ------- Phase C (WMMA): Vc = H_in@W_V, Qc/Kc = H_in@W_Q/W_K -------
    for (int tile = wave; tile < 108; tile += 32) {
      v8f acc = {0.f, 0.f, 0.f, 0.f, 0.f, 0.f, 0.f, 0.f};
      v16bf ah, al, bh, bl;
      if (tile < 96) {
        int m = tile >> 4, nt = tile & 15;
        const int abase = (m * BPW_ + row) * H_ + 8 * half;
        for (int kt = 0; kt < 8; ++kt) {
          ah = load_a_stage(sHh, abase + kt * 32);
          al = load_a_stage(sHl, abase + kt * 32);
          load_b_tile(ws, SLOT_WV + m * 128 + kt * 16 + nt, lane, bh, bl);
          acc = wmma3(acc, ah, al, bh, bl);
        }
        int hcol = nt * 16 + nIdx;
#pragma unroll
        for (int r = 0; r < 8; ++r)
          sVc[((r + 8 * half) * M_ + m) * H_ + hcol] = acc[r];  // Vc[b][m][h]
      } else {
        int s = tile - 96;
        bool isK = (s >= 6);
        int m = isK ? s - 6 : s;
        const int abase = (m * BPW_ + row) * H_ + 8 * half;
        for (int kt = 0; kt < 8; ++kt) {
          ah = load_a_stage(sHh, abase + kt * 32);
          al = load_a_stage(sHl, abase + kt * 32);
          load_b_tile(ws, (isK ? SLOT_WK : SLOT_WQ) + m * 8 + kt, lane, bh, bl);
          acc = wmma3(acc, ah, al, bh, bl);
        }
        float* dst = isK ? sKc : sQc;
#pragma unroll
        for (int r = 0; r < 8; ++r)
          dst[((r + 8 * half) * M_ + m) * AK_ + nIdx] = acc[r];
      }
    }
    __syncthreads();

    // ---------------- Phase D (VALU): 6x6 attention softmax ----------------
    if (tid < BPW_ * M_) {
      int b = tid / M_, m = tid % M_;
      const float scale = 0.25f;  // 1/sqrt(16)
      float sc[M_], mx = -1e30f;
      for (int n = 0; n < M_; ++n) {
        float s = 0.f;
        for (int k = 0; k < AK_; ++k)
          s += sQc[(b * M_ + m) * AK_ + k] * sKc[(b * M_ + n) * AK_ + k];
        sc[n] = s * scale;
        mx = fmaxf(mx, sc[n]);
      }
      float den = 0.f;
      for (int n = 0; n < M_; ++n) { sc[n] = __expf(sc[n] - mx); den += sc[n]; }
      float inv = 1.f / den;
      for (int n = 0; n < M_; ++n) {
        float a = sc[n] * inv;
        sAtt[(b * M_ + m) * M_ + n] = a;
        __builtin_nontemporal_store(
            a, &out[ATTN_OFF + (((long)t * B_ + wg * BPW_ + b) * M_ + m) * M_ + n]);
      }
    }
    __syncthreads();

    // -------- Phase E (VALU): H_new = H_in + attn-mix(Vc); stream out --------
    for (int i = tid; i < BPW_ * M_ * H_; i += NTHREADS_) {
      int b = i / (M_ * H_), n = (i >> 8) % M_, h = i & 255;
      float comm = 0.f;
#pragma unroll
      for (int m = 0; m < M_; ++m)
        comm += sAtt[(b * M_ + m) * M_ + n] * sVc[(b * M_ + m) * H_ + h];
      float hn = sH[(b * M_ + n) * H_ + h] + comm;
      sH[(b * M_ + n) * H_ + h] = hn;
      long o = (((long)t * B_ + wg * BPW_ + b) * M_ + n) * H_ + h;
      __builtin_nontemporal_store(hn,   &out[HSEQ_OFF + o]);
      __builtin_nontemporal_store(comm, &out[ACOMM_OFF + o]);
    }
    __syncthreads();

    // ------- Phase F (WMMA): out = H_new[:,4:6].reshape(.,512)@W_out + b -------
    if (wave < 4) {
      int nt = wave;
      v8f acc = {0.f, 0.f, 0.f, 0.f, 0.f, 0.f, 0.f, 0.f};
      v16bf ah, al, bh, bl;
      for (int kt = 0; kt < 16; ++kt) {
        gather_a(lane, [&](int rw, int kk) {
          int K = kt * 32 + kk;
          return sH[(rw * M_ + 4 + (K >> 8)) * H_ + (K & 255)]; }, ah, al);
        load_b_tile(ws, SLOT_WOUT + kt * 4 + nt, lane, bh, bl);
        acc = wmma3(acc, ah, al, bh, bl);
      }
      int o = nt * 16 + nIdx;
      float bo = b_out[o];
#pragma unroll
      for (int r = 0; r < 8; ++r) {
        int bg = wg * BPW_ + r + 8 * half;
        out[OUTS_OFF + ((long)bg * T_ + t) * OUT_ + o] = acc[r] + bo;
      }
    }
    __syncthreads();
  }
}

// ---------------------------------------------------------------------------
extern "C" void kernel_launch(void* const* d_in, const int* in_sizes, int n_in,
                              void* d_out, int out_size, void* d_ws,
                              size_t ws_size, hipStream_t stream) {
  (void)in_sizes; (void)n_in; (void)out_size; (void)ws_size;
  const float* x         = (const float*)d_in[0];
  const float* W_ip      = (const float*)d_in[1];
  const float* b_ip      = (const float*)d_in[2];
  const float* U         = (const float*)d_in[3];
  const float* Vr        = (const float*)d_in[4];
  const float* cell_bias = (const float*)d_in[5];
  const float* Wtask     = (const float*)d_in[6];
  const float* Wsens     = (const float*)d_in[7];
  const float* W_Q       = (const float*)d_in[8];
  const float* W_K       = (const float*)d_in[9];
  const float* W_V       = (const float*)d_in[10];
  const float* W_out     = (const float*)d_in[11];
  const float* b_out     = (const float*)d_in[12];
  float*  out = (float*)d_out;
  __bf16* ws  = (__bf16*)d_ws;   // needs 1,835,008 bf16 = 3.5 MB

  rim_prep_kernel<<<dim3(N_SLOTS), dim3(512), 0, stream>>>(
      W_ip, W_Q, W_K, W_V, Vr, W_out, ws);

  rim_main_kernel<<<dim3(B_ / BPW_), dim3(NTHREADS_), SMEM_BYTES, stream>>>(
      x, U, b_ip, cell_bias, Wtask, Wsens, b_out, ws, out);
}
